// ABHUE_61555471286966
// MI455X (gfx1250) — compile-verified, matrix operand order checked by reading, not run on hardware
//
#include <hip/hip_runtime.h>

// Problem constants (fixed by the reference)
#define S_    1025
#define W_    128
#define H_    200
#define G4    800          // 4*H
#define KP    224          // K padded to 7*32
#define KC    7            // K chunks of 32
#define NTILES 50          // G4/16
#define MID   512
#define TSEQ  513          // mid+1

typedef __attribute__((ext_vector_type(8)))  float  v8f;
typedef __attribute__((ext_vector_type(16))) __bf16 v16bf;

union AB16 { v16bf bf; uint4 q[2]; unsigned int u[8]; };

// Workspace layout (bytes, 16B aligned)
#define OFF_BW     0u              // 6*7*50*256 dwords = 2,150,400 B
#define OFF_BIASC  2150400u       // 800 f32
#define OFF_BIAST  2153600u       // 800 f32
#define OFF_EMB    2156800u       // 1025*200 f32 = 820,000 B
#define OFF_GXE    2976800u       // 2*513*800 f32 = 3,283,200 B
#define OFF_PP     6260000u       // 2*200 f32

__device__ __forceinline__ unsigned int f2bf(float x) {
    unsigned int u = __float_as_uint(x);
    return (u + 0x7FFFu + ((u >> 16) & 1u)) >> 16;   // RNE
}
__device__ __forceinline__ float sigf(float x) {
    return 1.0f / (1.0f + __expf(-x));
}
__device__ __forceinline__ float tanh_f(float x) {
    x = fminf(15.0f, fmaxf(-15.0f, x));
    float e = __expf(2.0f * x);
    return (e - 1.0f) / (e + 1.0f);
}

// ---- WMMA helpers ----------------------------------------------------------
// A operand: 16x32 bf16 tile from row-major LDS [rows][KP] (ushort). Lane L:
// m = L&15, half = L>>4; regs 0..3 = K kc*32+8h+{0..7} (one b128),
// regs 4..7 = K kc*32+16+8h+{0..7} (one b128).
__device__ __forceinline__ v16bf loadA(const unsigned short* aBase, int kc) {
    AB16 A;
    const uint4* pa = (const uint4*)(aBase + kc * 32);
    A.q[0] = pa[0];
    A.q[1] = pa[2];
    return A.bf;
}
// B operand: pre-packed per-lane register image, 256 dwords per (mat,kc,nt) tile.
__device__ __forceinline__ v16bf loadB(const unsigned int* __restrict__ bw, int off) {
    AB16 B;
    const uint4* pb = (const uint4*)(bw + off);
    B.q[0] = pb[0];
    B.q[1] = pb[1];
    return B.bf;
}
__device__ __forceinline__ v8f wmma_r(v16bf a, v16bf b, v8f acc) {
    return __builtin_amdgcn_wmma_f32_16x16x32_bf16(
        false, a, false, b, (short)0, acc, false, false);
}
__device__ __forceinline__ void store_tile(float* gate, const float* biasS,
                                           v8f a, int n, int half, int rowOff) {
    float bsv = biasS[n];
#pragma unroll
    for (int r = 0; r < 8; ++r) gate[(rowOff + r + 8 * half) * G4 + n] = a[r] + bsv;
}

// ---- Kernel 1: pack weights into WMMA B-register images + combined biases --
// mats: 0=wih_ctx 1=whh_ctx 2=wih_tgt 3=whh_tgt 4=whh_prev 5=whh_post
__global__ void prep_kernel(const float* __restrict__ wih_ctx, const float* __restrict__ whh_ctx,
                            const float* __restrict__ wih_tgt, const float* __restrict__ whh_tgt,
                            const float* __restrict__ whh_prev, const float* __restrict__ whh_post,
                            const float* __restrict__ bih_ctx, const float* __restrict__ bhh_ctx,
                            const float* __restrict__ bih_tgt, const float* __restrict__ bhh_tgt,
                            unsigned int* __restrict__ bw,
                            float* __restrict__ biasC, float* __restrict__ biasT) {
    int idx = blockIdx.x * blockDim.x + threadIdx.x;
    if (idx < G4)            biasC[idx]      = bih_ctx[idx]      + bhh_ctx[idx];
    else if (idx < 2 * G4)   biasT[idx - G4] = bih_tgt[idx - G4] + bhh_tgt[idx - G4];
    const int TOT = 6 * KC * NTILES * 256;
    if (idx >= TOT) return;
    int j = idx & 7, lane = (idx >> 3) & 31;
    int t2 = idx >> 8;
    int nt = t2 % NTILES;
    int t3 = t2 / NTILES;
    int kc = t3 % KC;
    int mat = t3 / KC;
    const float* Wm;
    switch (mat) {
        case 0: Wm = wih_ctx; break;
        case 1: Wm = whh_ctx; break;
        case 2: Wm = wih_tgt; break;
        case 3: Wm = whh_tgt; break;
        case 4: Wm = whh_prev; break;
        default: Wm = whh_post; break;
    }
    int n = lane & 15, half = lane >> 4;
    int k0 = (j < 4) ? (8 * half + 2 * j) : (16 + 8 * half + 2 * (j - 4));
    int kg = kc * 32 + k0;
    int ng = nt * 16 + n;
    float v0 = (kg     < H_) ? Wm[ng * H_ + kg]     : 0.0f;
    float v1 = (kg + 1 < H_) ? Wm[ng * H_ + kg + 1] : 0.0f;
    bw[idx] = f2bf(v0) | (f2bf(v1) << 16);
}

// ---- Kernel 2: batched word-level LSTM, M=32 sentences per WG --------------
// WGs 0..32 = ctx sentences, WG 33 = target LSTM (all rows = sentence MID).
__global__ __launch_bounds__(320) void ctx_kernel(
    const float* __restrict__ words, const int* __restrict__ lengths,
    const float* __restrict__ h0w, const float* __restrict__ c0w,
    const unsigned int* __restrict__ bw,
    const float* __restrict__ biasC, const float* __restrict__ biasT,
    float* __restrict__ emb) {
    extern __shared__ char smem[];
    unsigned short* xA   = (unsigned short*)(smem);            // [32][KP] bf16
    unsigned short* hA   = (unsigned short*)(smem + 14336);    // [32][KP] bf16
    float*          gate = (float*)(smem + 28672);             // [32][G4]
    float*          cS   = (float*)(smem + 131072);            // [32][H_]
    float*          biasS= (float*)(smem + 156672);            // [G4]
    int*            lenS = (int*)(smem + 159872);              // [32]

    const int tid = threadIdx.x;
    const int lane = tid & 31, wv = tid >> 5;
    const int wg = blockIdx.x;
    const bool isTgt = (wg == 33);
    const int s0 = isTgt ? MID : wg * 32;
    const int m_ = lane & 15, half = lane >> 4;
    const int matW0 = isTgt ? 2 : 0;      // wih
    const int matW1 = matW0 + 1;          // whh
    const float* bias = isTgt ? biasT : biasC;

    for (int e = tid; e < G4; e += 320) biasS[e] = bias[e];
    if (tid < 32) {
        int s = isTgt ? MID : min(s0 + tid, S_ - 1);
        lenS[tid] = lengths[s];
    }
    for (int e = tid; e < 32 * KP; e += 320) {
        int m = e / KP, k = e % KP;
        int s = isTgt ? MID : min(s0 + m, S_ - 1);
        float hv = (k < H_) ? h0w[s * H_ + k] : 0.0f;
        hA[e] = (unsigned short)f2bf(hv);
        xA[e] = 0;                         // pad columns stay zero forever
    }
    for (int e = tid; e < 32 * H_; e += 320) {
        int m = e / H_, k = e % H_;
        int s = isTgt ? MID : min(s0 + m, S_ - 1);
        cS[e] = c0w[s * H_ + k];
    }

    const unsigned short* aX0 = &xA[m_ * KP + 8 * half];
    const unsigned short* aX1 = &xA[(16 + m_) * KP + 8 * half];
    const unsigned short* aH0 = &hA[m_ * KP + 8 * half];
    const unsigned short* aH1 = &hA[(16 + m_) * KP + 8 * half];

    int tb = 0;   // always 0, but opaque to the compiler: blocks LICM of the
                  // loop-invariant weight loads without losing the global
                  // address space (keeps global_load_b128 + LOADcnt only)

    for (int t = 0; t < W_; ++t) {
        // stage x_t (bf16) into LDS; words is streamed once -> non-temporal
        for (int e = tid; e < 32 * H_; e += 320) {
            int m = e / H_, k = e % H_;
            int s = isTgt ? MID : min(s0 + m, S_ - 1);
            float xv = __builtin_nontemporal_load(&words[(s * W_ + t) * H_ + k]);
            xA[m * KP + k] = (unsigned short)f2bf(xv);
        }
        __syncthreads();

        v8f p0 = {0}, p1 = {0}, p2 = {0}, p3 = {0}, p4 = {0};   // rows 0..15
        v8f q0 = {0}, q1 = {0}, q2 = {0}, q3 = {0}, q4 = {0};   // rows 16..31
#pragma unroll
        for (int kc = 0; kc < KC; ++kc) {                       // x @ wih^T
            asm volatile("" : "+s"(tb));    // block cross-iteration hoisting
            v16bf a0 = loadA(aX0, kc);
            v16bf a1 = loadA(aX1, kc);
            int base = ((matW0 * KC + kc) * NTILES + wv * 5) * 256 + lane * 8 + tb;
            v16bf b;
            b = loadB(bw, base);        p0 = wmma_r(a0, b, p0); q0 = wmma_r(a1, b, q0);
            b = loadB(bw, base + 256);  p1 = wmma_r(a0, b, p1); q1 = wmma_r(a1, b, q1);
            b = loadB(bw, base + 512);  p2 = wmma_r(a0, b, p2); q2 = wmma_r(a1, b, q2);
            b = loadB(bw, base + 768);  p3 = wmma_r(a0, b, p3); q3 = wmma_r(a1, b, q3);
            b = loadB(bw, base + 1024); p4 = wmma_r(a0, b, p4); q4 = wmma_r(a1, b, q4);
        }
#pragma unroll
        for (int kc = 0; kc < KC; ++kc) {                       // h @ whh^T
            asm volatile("" : "+s"(tb));
            v16bf a0 = loadA(aH0, kc);
            v16bf a1 = loadA(aH1, kc);
            int base = ((matW1 * KC + kc) * NTILES + wv * 5) * 256 + lane * 8 + tb;
            v16bf b;
            b = loadB(bw, base);        p0 = wmma_r(a0, b, p0); q0 = wmma_r(a1, b, q0);
            b = loadB(bw, base + 256);  p1 = wmma_r(a0, b, p1); q1 = wmma_r(a1, b, q1);
            b = loadB(bw, base + 512);  p2 = wmma_r(a0, b, p2); q2 = wmma_r(a1, b, q2);
            b = loadB(bw, base + 768);  p3 = wmma_r(a0, b, p3); q3 = wmma_r(a1, b, q3);
            b = loadB(bw, base + 1024); p4 = wmma_r(a0, b, p4); q4 = wmma_r(a1, b, q4);
        }
        store_tile(gate, biasS, p0, wv * 80 +  0 + m_, half, 0);
        store_tile(gate, biasS, p1, wv * 80 + 16 + m_, half, 0);
        store_tile(gate, biasS, p2, wv * 80 + 32 + m_, half, 0);
        store_tile(gate, biasS, p3, wv * 80 + 48 + m_, half, 0);
        store_tile(gate, biasS, p4, wv * 80 + 64 + m_, half, 0);
        store_tile(gate, biasS, q0, wv * 80 +  0 + m_, half, 16);
        store_tile(gate, biasS, q1, wv * 80 + 16 + m_, half, 16);
        store_tile(gate, biasS, q2, wv * 80 + 32 + m_, half, 16);
        store_tile(gate, biasS, q3, wv * 80 + 48 + m_, half, 16);
        store_tile(gate, biasS, q4, wv * 80 + 64 + m_, half, 16);
        __syncthreads();

        // elementwise LSTM update (torch gate order i,f,g,o)
        for (int e = tid; e < 32 * H_; e += 320) {
            int m = e / H_, k = e % H_;
            float gi = sigf(gate[m * G4 + k]);
            float gf = sigf(gate[m * G4 + H_ + k]);
            float gg = tanh_f(gate[m * G4 + 2 * H_ + k]);
            float go = sigf(gate[m * G4 + 3 * H_ + k]);
            float c  = gf * cS[e] + gi * gg;
            cS[e] = c;
            float h = go * tanh_f(c);
            hA[m * KP + k] = (unsigned short)f2bf(h);
            if (lenS[m] - 1 == t) {
                if (isTgt) {
                    if (m == 0) emb[MID * H_ + k] = h;
                } else {
                    int s = s0 + m;
                    if (s < S_ && s != MID) emb[s * H_ + k] = h;
                }
            }
        }
        __syncthreads();
    }
}

// ---- Kernel 3: input projections for the two sentence-level chains ---------
__global__ void embproj_kernel(const float* __restrict__ emb,
                               const float* __restrict__ wih_prev,
                               const float* __restrict__ bih_prev, const float* __restrict__ bhh_prev,
                               const float* __restrict__ wih_post,
                               const float* __restrict__ bih_post, const float* __restrict__ bhh_post,
                               float* __restrict__ gxe) {
    int idx = blockIdx.x * blockDim.x + threadIdx.x;
    const int TOT = 2 * TSEQ * G4;
    if (idx >= TOT) return;
    int sq = idx / (TSEQ * G4);
    int rem = idx - sq * (TSEQ * G4);
    int t = rem / G4, n = rem % G4;
    int row = sq ? (S_ - 1 - t) : t;                       // post walks reversed
    const float* w = (sq ? wih_post : wih_prev) + n * H_;
    const float* er = emb + row * H_;
    float a = sq ? (bih_post[n] + bhh_post[n]) : (bih_prev[n] + bhh_prev[n]);
    for (int k = 0; k < H_; ++k) a = fmaf(er[k], w[k], a);
    gxe[idx] = a;
}

// ---- Kernel 4: sequential prev/post chains; whh persistent in VGPRs --------
// min-waves=1: allow >256 VGPRs/wave so the 280-reg weight image never spills.
__global__ __launch_bounds__(320, 1) void seq_kernel(
    const float* __restrict__ h0p, const float* __restrict__ c0p,
    const float* __restrict__ h0q, const float* __restrict__ c0q,
    const unsigned int* __restrict__ bw, const float* __restrict__ gxe,
    float* __restrict__ pp) {
    __shared__ __align__(16) unsigned short hA[16 * KP];   // only row 0 live
    __shared__ float gbuf[G4];

    const int tid = threadIdx.x;
    const int lane = tid & 31, wv = tid >> 5;
    const int b = blockIdx.x;                              // 0=prev, 1=post
    const float* h0 = b ? h0q : h0p;
    const float* c0 = b ? c0q : c0p;
    const int mat = 4 + b;
    const float* gx = gxe + b * (TSEQ * G4);
    const int m_ = lane & 15, half = lane >> 4;

    // persistent B: wave's 5 N-tiles x 7 K-chunks (280 VGPRs, stays resident)
    v16bf Bv[KC * 5];
#pragma unroll
    for (int kc = 0; kc < KC; ++kc) {
        int base = ((mat * KC + kc) * NTILES + wv * 5) * 256 + lane * 8;
#pragma unroll
        for (int nt = 0; nt < 5; ++nt) Bv[kc * 5 + nt] = loadB(bw, base + nt * 256);
    }

    for (int e = tid; e < 16 * KP; e += 320) {
        int m = e / KP, k = e % KP;
        hA[e] = (m == 0 && k < H_) ? (unsigned short)f2bf(h0[k]) : 0;
    }
    float cr = (tid < H_) ? c0[tid] : 0.0f;
    float hlast = 0.0f;
    const unsigned short* aBase = &hA[m_ * KP + 8 * half];

    for (int t = 0; t < TSEQ; ++t) {
        __syncthreads();                                   // hA ready
        v8f acc0 = {0}, acc1 = {0}, acc2 = {0}, acc3 = {0}, acc4 = {0};
#pragma unroll
        for (int kc = 0; kc < KC; ++kc) {
            v16bf a = loadA(aBase, kc);
            acc0 = wmma_r(a, Bv[kc * 5 + 0], acc0);
            acc1 = wmma_r(a, Bv[kc * 5 + 1], acc1);
            acc2 = wmma_r(a, Bv[kc * 5 + 2], acc2);
            acc3 = wmma_r(a, Bv[kc * 5 + 3], acc3);
            acc4 = wmma_r(a, Bv[kc * 5 + 4], acc4);
        }
        if (half == 0) {                                   // D row 0 = reg0, lanes 0-15
            int n = wv * 80 + m_;
            gbuf[n]      = acc0[0] + gx[t * G4 + n];
            gbuf[n + 16] = acc1[0] + gx[t * G4 + n + 16];
            gbuf[n + 32] = acc2[0] + gx[t * G4 + n + 32];
            gbuf[n + 48] = acc3[0] + gx[t * G4 + n + 48];
            gbuf[n + 64] = acc4[0] + gx[t * G4 + n + 64];
        }
        __syncthreads();
        if (tid < H_) {
            float gi = sigf(gbuf[tid]);
            float gf = sigf(gbuf[H_ + tid]);
            float gg = tanh_f(gbuf[2 * H_ + tid]);
            float go = sigf(gbuf[3 * H_ + tid]);
            cr = gf * cr + gi * gg;
            hlast = go * tanh_f(cr);
            hA[tid] = (unsigned short)f2bf(hlast);         // row 0
        }
    }
    __syncthreads();
    if (tid < H_) pp[b * H_ + tid] = hlast;
}

// ---- Kernel 5: final FC ----------------------------------------------------
__global__ void fc_kernel(const float* __restrict__ pp, const float* __restrict__ fc_w,
                          const float* __restrict__ fc_b, float* __restrict__ out) {
    int j = blockIdx.x * blockDim.x + threadIdx.x;
    if (j >= H_) return;
    float a = fc_b[j];
    const float* w = fc_w + j * (2 * H_);
    for (int k = 0; k < 2 * H_; ++k) a = fmaf(pp[k], w[k], a);
    out[j] = a;
}

extern "C" void kernel_launch(void* const* d_in, const int* in_sizes, int n_in,
                              void* d_out, int out_size, void* d_ws, size_t ws_size,
                              hipStream_t stream) {
    (void)in_sizes; (void)n_in; (void)out_size; (void)ws_size;
    const float* words   = (const float*)d_in[0];
    const int*   lengths = (const int*)d_in[1];
    const float* h0w = (const float*)d_in[2];
    const float* c0w = (const float*)d_in[3];
    const float* h0p = (const float*)d_in[4];
    const float* c0p = (const float*)d_in[5];
    const float* h0q = (const float*)d_in[6];
    const float* c0q = (const float*)d_in[7];
    const float* wih_ctx = (const float*)d_in[8];
    const float* whh_ctx = (const float*)d_in[9];
    const float* bih_ctx = (const float*)d_in[10];
    const float* bhh_ctx = (const float*)d_in[11];
    const float* wih_tgt = (const float*)d_in[12];
    const float* whh_tgt = (const float*)d_in[13];
    const float* bih_tgt = (const float*)d_in[14];
    const float* bhh_tgt = (const float*)d_in[15];
    const float* wih_prev = (const float*)d_in[16];
    const float* whh_prev = (const float*)d_in[17];
    const float* bih_prev = (const float*)d_in[18];
    const float* bhh_prev = (const float*)d_in[19];
    const float* wih_post = (const float*)d_in[20];
    const float* whh_post = (const float*)d_in[21];
    const float* bih_post = (const float*)d_in[22];
    const float* bhh_post = (const float*)d_in[23];
    const float* fc_w = (const float*)d_in[24];
    const float* fc_b = (const float*)d_in[25];

    char* ws = (char*)d_ws;
    unsigned int* bw    = (unsigned int*)(ws + OFF_BW);
    float*        biasC = (float*)(ws + OFF_BIASC);
    float*        biasT = (float*)(ws + OFF_BIAST);
    float*        emb   = (float*)(ws + OFF_EMB);
    float*        gxe   = (float*)(ws + OFF_GXE);
    float*        pp    = (float*)(ws + OFF_PP);

    // 6*7*50*256 = 537600 dwords -> 2100 blocks of 256
    prep_kernel<<<2100, 256, 0, stream>>>(wih_ctx, whh_ctx, wih_tgt, whh_tgt,
                                          whh_prev, whh_post,
                                          bih_ctx, bhh_ctx, bih_tgt, bhh_tgt,
                                          bw, biasC, biasT);
    // 33 ctx WGs (32 sentences each) + 1 target WG; 160000 B dynamic LDS
    ctx_kernel<<<34, 320, 160000, stream>>>(words, lengths, h0w, c0w,
                                            bw, biasC, biasT, emb);
    embproj_kernel<<<(2 * TSEQ * G4 + 255) / 256, 256, 0, stream>>>(
        emb, wih_prev, bih_prev, bhh_prev, wih_post, bih_post, bhh_post, gxe);
    seq_kernel<<<2, 320, 0, stream>>>(h0p, c0p, h0q, c0q, bw, gxe, pp);
    fc_kernel<<<1, 256, 0, stream>>>(pp, fc_w, fc_b, (float*)d_out);
}
